// TopKAttention_28140625723861
// MI455X (gfx1250) — matrix operand order repaired and checked by reading, not compiled, not run on hardware
//
#include <hip/hip_runtime.h>

#define B_    2
#define H_    16
#define S_    2048
#define D_    128
#define KTOP  32
#define QTILE 16
#define CHUNK 256                 // keys per chunk = 8 waves * 2 tiles * 16
#define NCHUNK (S_ / CHUNK)       // 8
#define NELEM  (B_ * H_ * S_ * D_)  // 8388608 per tensor

typedef __attribute__((ext_vector_type(16))) __bf16          v16bf;
typedef __attribute__((ext_vector_type(16))) unsigned short  v16us;
typedef __attribute__((ext_vector_type(8)))  unsigned short  v8us;
typedef __attribute__((ext_vector_type(8)))  float           v8f;
typedef __attribute__((ext_vector_type(4)))  float           f4;

// fp32 -> bf16 round-to-nearest-even (used only in the memory-bound pre-pass)
__device__ __forceinline__ unsigned short f2bf(float f) {
    unsigned int u = __builtin_bit_cast(unsigned int, f);
    u += 0x7FFFu + ((u >> 16) & 1u);
    return (unsigned short)(u >> 16);
}

// ---------------- pre-pass: fp32 -> bf16, 8 elements / thread ----------------
__global__ __launch_bounds__(256)
void cvt_bf16_kernel(const float* __restrict__ src, unsigned short* __restrict__ dst, int n8) {
    const int i = blockIdx.x * 256 + threadIdx.x;
    if (i >= n8) return;
    const f4* s = (const f4*)src + (size_t)i * 2;
    const f4 x0 = s[0], x1 = s[1];
    v8us o;
    o[0] = f2bf(x0.x); o[1] = f2bf(x0.y); o[2] = f2bf(x0.z); o[3] = f2bf(x0.w);
    o[4] = f2bf(x1.x); o[5] = f2bf(x1.y); o[6] = f2bf(x1.z); o[7] = f2bf(x1.w);
    *((v8us*)dst + i) = o;
}

// -------- operand loads from pre-converted bf16 (pure b128 loads, no VALU) --------
// A (16x32 MxK): row M=lane&15, K = d0..d0+15, d0 = kk + hi*16
__device__ __forceinline__ v16bf load_a16(const unsigned short* __restrict__ qr, int kk, int hi) {
    const int d0 = kk + hi * 16;
    const v16us u = *(const v16us*)(qr + d0);
    return __builtin_bit_cast(v16bf, u);
}
// B (32x16 KxN): lane holds column N=lane&15; lane<16 -> K=0..15, lane>=16 -> K=16..31
__device__ __forceinline__ v16bf load_b16(const unsigned short* __restrict__ kr, int off) {
    return __builtin_bit_cast(v16bf, *(const v16us*)(kr + off));
}

// 8 B-operands (2 key tiles x 4 K-steps) held in distinct registers
struct BTiles { v16bf b[8]; };

__device__ __forceinline__ void load_btiles(BTiles& B,
        const unsigned short* __restrict__ Kb, size_t base,
        int ch, int wave, int lan15, int boff) {
    const int kb0 = ch * CHUNK + wave * 32;
    const unsigned short* kr0 = Kb + base + (size_t)(kb0 + lan15) * D_;
    const unsigned short* kr1 = kr0 + 16 * D_;
#pragma unroll
    for (int s = 0; s < 4; ++s) {
        B.b[s]     = load_b16(kr0, s * 32 + boff);
        B.b[4 + s] = load_b16(kr1, s * 32 + boff);
    }
}

__global__ __launch_bounds__(256, 1)
void topk_attn_kernel(const unsigned short* __restrict__ Qb,
                      const unsigned short* __restrict__ Kb,
                      const float* __restrict__ V, float* __restrict__ Out) {
    __shared__ float sc[QTILE][CHUNK];   // 16 KB score staging
    __shared__ float ts[QTILE][KTOP];    // 2 KB top-k scores
    __shared__ int   ti[QTILE][KTOP];    // 2 KB top-k indices
    __shared__ float pr[QTILE][KTOP];    // 2 KB softmax probs

    const int qt = blockIdx.x;                    // query tile 0..127
    const int bh = blockIdx.y;                    // 0..31
    const size_t base = (size_t)bh * S_ * D_;
    const int q0 = qt * QTILE;

    const int t     = threadIdx.x;
    const int wave  = t >> 5;
    const int lane  = t & 31;
    const int hi    = lane >> 4;
    const int lan15 = lane & 15;
    const int boff  = hi * 16;                    // B-operand K-half select

    for (int i = t; i < QTILE * KTOP; i += 256) {
        ((float*)ts)[i] = -3.0e38f;
        ((int*)ti)[i]   = 0;
    }
    __syncthreads();

    // Q tile A-operands: loaded once as bf16, reused for all 2048 keys
    const unsigned short* qrow = Qb + base + (size_t)(q0 + lan15) * D_;
    v16bf A[4];
    A[0] = load_a16(qrow,  0, hi);
    A[1] = load_a16(qrow, 32, hi);
    A[2] = load_a16(qrow, 64, hi);
    A[3] = load_a16(qrow, 96, hi);

    float minv = -3.0e38f;   // top-k running min (valid in threads t<16)
    int   minpos = 0;

    // WMMA burst + LDS stage + top-k update for one chunk's worth of scores.
    // B operands for the FOLLOWING chunk are already in flight when this runs,
    // so their latency hides behind the store/barrier/update phase.
#define PROCESS(BT, CH)                                                                              \
    do {                                                                                             \
        v8f c0 = {0.f, 0.f, 0.f, 0.f, 0.f, 0.f, 0.f, 0.f};                                           \
        v8f c1 = {0.f, 0.f, 0.f, 0.f, 0.f, 0.f, 0.f, 0.f};                                           \
        c0 = __builtin_amdgcn_wmma_f32_16x16x32_bf16(false, A[0], false, (BT).b[0], (short)0, c0, false, false); \
        c0 = __builtin_amdgcn_wmma_f32_16x16x32_bf16(false, A[1], false, (BT).b[1], (short)0, c0, false, false); \
        c0 = __builtin_amdgcn_wmma_f32_16x16x32_bf16(false, A[2], false, (BT).b[2], (short)0, c0, false, false); \
        c0 = __builtin_amdgcn_wmma_f32_16x16x32_bf16(false, A[3], false, (BT).b[3], (short)0, c0, false, false); \
        c1 = __builtin_amdgcn_wmma_f32_16x16x32_bf16(false, A[0], false, (BT).b[4], (short)0, c1, false, false); \
        c1 = __builtin_amdgcn_wmma_f32_16x16x32_bf16(false, A[1], false, (BT).b[5], (short)0, c1, false, false); \
        c1 = __builtin_amdgcn_wmma_f32_16x16x32_bf16(false, A[2], false, (BT).b[6], (short)0, c1, false, false); \
        c1 = __builtin_amdgcn_wmma_f32_16x16x32_bf16(false, A[3], false, (BT).b[7], (short)0, c1, false, false); \
        const int col0 = wave * 32 + lan15;                                                          \
        _Pragma("unroll")                                                                            \
        for (int r = 0; r < 8; ++r) {                                                                \
            sc[r + hi * 8][col0]      = c0[r];                                                       \
            sc[r + hi * 8][col0 + 16] = c1[r];                                                       \
        }                                                                                            \
        __syncthreads();                                                                             \
        if (t < QTILE) {                                                                             \
            const int q = t;                                                                         \
            for (int j = 0; j < CHUNK; ++j) {                                                        \
                const float s = sc[q][j];                                                            \
                if (s > minv) {                                                                      \
                    ts[q][minpos] = s;                                                               \
                    ti[q][minpos] = (CH) * CHUNK + j;                                                \
                    float m = ts[q][0]; int mp = 0;                                                  \
                    _Pragma("unroll")                                                                \
                    for (int k2 = 1; k2 < KTOP; ++k2) {                                              \
                        const float v2 = ts[q][k2];                                                  \
                        if (v2 < m) { m = v2; mp = k2; }                                             \
                    }                                                                                \
                    minv = m; minpos = mp;                                                           \
                }                                                                                    \
            }                                                                                        \
        }                                                                                            \
        __syncthreads();                                                                             \
    } while (0)

    BTiles Bcur, Bnxt;
    load_btiles(Bcur, Kb, base, 0, wave, lan15, boff);
    for (int ch = 0; ch < NCHUNK; ch += 2) {
        // issue next chunk's loads before touching this chunk's operands
        load_btiles(Bnxt, Kb, base, ch + 1, wave, lan15, boff);
        __builtin_prefetch(Kb + base + (size_t)((ch + 2) * CHUNK + wave * 32 + lan15) * D_, 0, 1);
        PROCESS(Bcur, ch);
        if (ch + 2 < NCHUNK)
            load_btiles(Bcur, Kb, base, ch + 2, wave, lan15, boff);
        PROCESS(Bnxt, ch + 1);
    }
#undef PROCESS

    // softmax over the 32 survivors, one thread per query
    if (t < QTILE) {
        float mx = -3.0e38f;
#pragma unroll
        for (int j = 0; j < KTOP; ++j) mx = fmaxf(mx, ts[t][j]);
        float sum = 0.f;
#pragma unroll
        for (int j = 0; j < KTOP; ++j) {
            const float e = __expf(ts[t][j] - mx);
            pr[t][j] = e; sum += e;
        }
        const float inv = 1.f / sum;
#pragma unroll
        for (int j = 0; j < KTOP; ++j) pr[t][j] *= inv;
    }
    __syncthreads();

    // output: 16 threads per query, 8 dims each (coalesced 512B V-row reads)
    {
        const int q  = t >> 4;
        const int d0 = (t & 15) * 8;
        float acc[8] = {0.f, 0.f, 0.f, 0.f, 0.f, 0.f, 0.f, 0.f};
        for (int j = 0; j < KTOP; ++j) {
            const float p = pr[q][j];
            const float* vrow = V + base + (size_t)ti[q][j] * D_ + d0;
#pragma unroll
            for (int i = 0; i < 8; ++i) acc[i] += p * vrow[i];
        }
        float* orow = Out + base + (size_t)(q0 + q) * D_ + d0;
#pragma unroll
        for (int i = 0; i < 8; ++i) orow[i] = acc[i];
    }
}

extern "C" void kernel_launch(void* const* d_in, const int* in_sizes, int n_in,
                              void* d_out, int out_size, void* d_ws, size_t ws_size,
                              hipStream_t stream) {
    (void)in_sizes; (void)n_in; (void)out_size; (void)ws_size;
    const float* Q = (const float*)d_in[0];
    const float* K = (const float*)d_in[1];
    const float* V = (const float*)d_in[2];
    float* Out = (float*)d_out;

    // workspace: bf16 copies of Q and K (2 * 16.78 MB = 33.6 MB)
    unsigned short* Qb = (unsigned short*)d_ws;
    unsigned short* Kb = Qb + (size_t)NELEM;

    const int n8 = NELEM / 8;                 // 1,048,576 threads, 8 elems each
    cvt_bf16_kernel<<<n8 / 256, 256, 0, stream>>>(Q, Qb, n8);
    cvt_bf16_kernel<<<n8 / 256, 256, 0, stream>>>(K, Kb, n8);

    dim3 grid(S_ / QTILE, B_ * H_);
    topk_attn_kernel<<<grid, 256, 0, stream>>>(Qb, Kb, V, Out);
}
